// FFTAttention_17489106830080
// MI455X (gfx1250) — matrix operand order repaired and checked by reading, not compiled, not run on hardware
//
#include <hip/hip_runtime.h>

// ---------------------------------------------------------------------------
// FFT attention for MI455X (gfx1250, wave32, WMMA, TDM).
// prep-weights -> rfft(q,k,v) -> proj GEMMs (WMMA f16, 2x2 reg tiles) ->
// QK^T scores -> softmax (rows of 1025) -> ctx = P@V -> out proj
// (TDM-staged B in LDS) -> irfft
// ---------------------------------------------------------------------------

#define B_   16
#define L_   2048
#define C_   512
#define H_   8
#define FREQ 1025
#define FP   1056               // FREQ padded to 33 * 32
#define M2T  33                 // 32-row tiles along F
#define MT_PROJ   (M2T * 16)    // 32x32 tiles per batch, N=512
#define MT_SCORE  (M2T * M2T)   // per (b,h)
#define MT_CTX    (M2T * 2)     // per (b,h), N=64
#define OUT_MCHUNK 5            // ceil(33/8) wave-chunks per (b,tn)

typedef _Float16 h16;
typedef __attribute__((ext_vector_type(16))) _Float16 v16h;
typedef __attribute__((ext_vector_type(8)))  float    v8f;
typedef __attribute__((ext_vector_type(4)))  unsigned int u32x4;
typedef __attribute__((ext_vector_type(8)))  int i32x8;
typedef __attribute__((ext_vector_type(4)))  int i32x4;

union FragH { v16h v; h16 h[16]; float4 q[2]; };
union FragF { v8f  v; float f[8]; };

__device__ __forceinline__ v8f wmma_f16(v16h a, v16h b, v8f c) {
  return __builtin_amdgcn_wmma_f32_16x16x32_f16(false, a, false, b, (short)0, c,
                                                false, false);
}

__device__ __forceinline__ v16h frag_zero() {
  FragH u;
#pragma unroll
  for (int i = 0; i < 16; ++i) u.h[i] = (h16)0.f;
  return u.v;
}

// A fragment (16x32 f16), ISA 7.12.2 interleaved K-halves.
__device__ __forceinline__ v16h frag_ld_a(const h16* __restrict__ rowp, int k0, int lane) {
  FragH u;
  const h16* p = rowp + k0 + ((lane >> 4) << 3);
  u.q[0] = *reinterpret_cast<const float4*>(p);
  u.q[1] = *reinterpret_cast<const float4*>(p + 16);
  return u.v;
}

// B fragment (32x16 f16): lane -> column N=lane&15, ISA 7.12.2 interleaved K-halves.
__device__ __forceinline__ v16h frag_ld_b(const h16* rowp, int k0, int lane) {
  FragH u;
  const h16* p = rowp + k0 + ((lane >> 4) << 3);
  u.q[0] = *reinterpret_cast<const float4*>(p);
  u.q[1] = *reinterpret_cast<const float4*>(p + 16);
  return u.v;
}

__device__ __forceinline__ v16h frag_ld_b_guard(const h16* __restrict__ rowp, int k0,
                                                int kmax, int lane) {
  FragH u;
  int kb = k0 + ((lane >> 4) << 3);
#pragma unroll
  for (int i = 0; i < 16; ++i) {
    int k = kb + i + ((i >> 3) << 3);
    u.h[i] = (k < kmax) ? rowp[k] : (h16)0.f;
  }
  return u.v;
}

// A fragment from f32 source (softmaxed probs), converted to f16 on the fly.
__device__ __forceinline__ v16h frag_ld_a_f32(const float* __restrict__ rowp, int k0,
                                              int kmax, int lane) {
  FragH u;
  int kb = k0 + ((lane >> 4) << 3);
#pragma unroll
  for (int i = 0; i < 8; ++i) {
    int k = kb + i;
    u.h[i] = (h16)((k < kmax) ? rowp[k] : 0.f);
  }
#pragma unroll
  for (int i = 0; i < 8; ++i) {
    int k = kb + 16 + i;
    u.h[8 + i] = (h16)((k < kmax) ? rowp[k] : 0.f);
  }
  return u.v;
}

// ---------------------------------------------------------------------------
__global__ void prep_weights_kernel(const float* __restrict__ Wq, const float* __restrict__ Wk,
                                    const float* __restrict__ Wv, const float* __restrict__ Wo,
                                    h16* __restrict__ Wqc, h16* __restrict__ Wkc,
                                    h16* __restrict__ Wvc, h16* __restrict__ Wo16,
                                    float* __restrict__ wsq, float* __restrict__ wsk,
                                    float* __restrict__ wsv) {
  int ob = blockIdx.x;
  int t  = threadIdx.x;
  __shared__ float red[3][256];
  float sq = 0.f, sk = 0.f, sv = 0.f;
  for (int c = t; c < 512; c += 256) {
    size_t i = ((size_t)ob * 512 + c) * 3;
    Wqc[(size_t)ob * 1024 + c]       = (h16)Wq[i];
    Wqc[(size_t)ob * 1024 + 512 + c] = (h16)Wq[i + 1];
    sq += Wq[i + 2];
    Wkc[(size_t)ob * 1024 + c]       = (h16)Wk[i];
    Wkc[(size_t)ob * 1024 + 512 + c] = (h16)Wk[i + 1];
    sk += Wk[i + 2];
    Wvc[(size_t)ob * 1024 + c]       = (h16)Wv[i];
    Wvc[(size_t)ob * 1024 + 512 + c] = (h16)Wv[i + 1];
    sv += Wv[i + 2];
    Wo16[(size_t)ob * 512 + c]         = (h16)Wo[(size_t)ob * 512 + c];
    Wo16[(size_t)(ob + 512) * 512 + c] = (h16)Wo[(size_t)(ob + 512) * 512 + c];
  }
  red[0][t] = sq; red[1][t] = sk; red[2][t] = sv;
  __syncthreads();
  for (int o = 128; o > 0; o >>= 1) {
    if (t < o) {
      red[0][t] += red[0][t + o];
      red[1][t] += red[1][t + o];
      red[2][t] += red[2][t + o];
    }
    __syncthreads();
  }
  if (t == 0) { wsq[ob] = red[0][0]; wsk[ob] = red[1][0]; wsv[ob] = red[2][0]; }
}

// ---------------------------------------------------------------------------
// 1024-pt in-place complex radix-2 DIT FFT in LDS, 256 threads.
// ---------------------------------------------------------------------------
__device__ void fft1024(float* re, float* im, float sgn) {
#pragma unroll 1
  for (int s = 1; s <= 10; ++s) {
    int m = 1 << s, half = m >> 1;
    for (int t = (int)threadIdx.x; t < 512; t += 256) {
      int grp = t >> (s - 1);
      int k   = t & (half - 1);
      int i1  = (grp << s) + k;
      int i2  = i1 + half;
      float ang = sgn * 6.28318530718f * (float)k / (float)m;
      float c, sn;
      __sincosf(ang, &sn, &c);
      float vr = re[i2] * c - im[i2] * sn;
      float vi = re[i2] * sn + im[i2] * c;
      re[i2] = re[i1] - vr;
      im[i2] = im[i1] - vi;
      re[i1] += vr;
      im[i1] += vi;
    }
    __syncthreads();
  }
}

// ---------------------------------------------------------------------------
__global__ void rfft_feats_kernel(const float* __restrict__ x, h16* __restrict__ A) {
  __shared__ float re[1024], im[1024];
  int bc = blockIdx.x;
  int b = bc >> 9, c = bc & 511;
  const float* xb = x + (size_t)b * L_ * C_ + c;
  for (int j = threadIdx.x; j < 1024; j += 256) {
    int r = (int)(__brev((unsigned)j) >> 22);
    re[r] = xb[(size_t)(2 * j) * C_];
    im[r] = xb[(size_t)(2 * j + 1) * C_];
  }
  __syncthreads();
  fft1024(re, im, -1.f);

  h16* Ab = A + (size_t)b * FP * 1024;
  for (int k = threadIdx.x; k <= 1024; k += 256) {
    int ka  = k & 1023;
    int kb2 = (1024 - k) & 1023;
    float Sr = re[ka] + re[kb2], Si = im[ka] - im[kb2];
    float Dr = re[ka] - re[kb2], Di = im[ka] + im[kb2];
    float th = 3.14159265359f * (float)k / 1024.f;
    float ct, st;
    __sincosf(th, &st, &ct);
    float Xr = 0.5f * (Sr + (ct * Di - st * Dr));
    float Xi = 0.5f * (Si - (ct * Dr + st * Di));
    Ab[(size_t)k * 1024 + c]       = (h16)Xr;
    Ab[(size_t)k * 1024 + 512 + c] = (h16)Xi;
  }
  for (int f = FREQ + threadIdx.x; f < FP; f += 256) {
    Ab[(size_t)f * 1024 + c]       = (h16)0.f;
    Ab[(size_t)f * 1024 + 512 + c] = (h16)0.f;
  }
}

// ---------------------------------------------------------------------------
// Projection GEMM, 2x2 register tiles (32x32 per wave), K=1024.
// ---------------------------------------------------------------------------
__global__ void proj_gemm_kernel(const h16* __restrict__ A, const h16* __restrict__ W,
                                 const float* __restrict__ bias, const float* __restrict__ wsum,
                                 h16* __restrict__ outRM, h16* __restrict__ outTR,
                                 int transposed) {
  int lane = threadIdx.x & 31, wave = threadIdx.x >> 5;
  long tile = (long)blockIdx.x * 8 + wave;
  int b   = (int)(tile / MT_PROJ);
  int rem = (int)(tile % MT_PROJ);
  int tm = rem >> 4, tn = rem & 15;

  const h16* ar0 = A + (size_t)b * FP * 1024 + (size_t)(tm * 32 + (lane & 15)) * 1024;
  const h16* ar1 = ar0 + (size_t)16 * 1024;
  const h16* br0 = W + (size_t)(tn * 32 + (lane & 15)) * 1024;
  const h16* br1 = br0 + (size_t)16 * 1024;

  FragF acc[2][2];
#pragma unroll
  for (int i = 0; i < 2; ++i)
#pragma unroll
    for (int j = 0; j < 2; ++j)
#pragma unroll
      for (int r = 0; r < 8; ++r) acc[i][j].f[r] = 0.f;

#pragma unroll 4
  for (int k0 = 0; k0 < 1024; k0 += 32) {
    if (k0 + 32 < 1024) {
      __builtin_prefetch(ar0 + k0 + 32, 0, 1);
      __builtin_prefetch(br0 + k0 + 32, 0, 1);
    }
    v16h a0 = frag_ld_a(ar0, k0, lane);
    v16h a1 = frag_ld_a(ar1, k0, lane);
    v16h b0 = frag_ld_b(br0, k0, lane);
    v16h b1 = frag_ld_b(br1, k0, lane);
    acc[0][0].v = wmma_f16(a0, b0, acc[0][0].v);
    acc[0][1].v = wmma_f16(a0, b1, acc[0][1].v);
    acc[1][0].v = wmma_f16(a1, b0, acc[1][0].v);
    acc[1][1].v = wmma_f16(a1, b1, acc[1][1].v);
  }

  int lhi = lane >> 4, lcol = lane & 15;
#pragma unroll
  for (int j = 0; j < 2; ++j) {
    int gcol = tn * 32 + j * 16 + lcol;
    float bn = bias[gcol], wz = wsum[gcol];
#pragma unroll
    for (int i = 0; i < 2; ++i) {
#pragma unroll
      for (int r = 0; r < 8; ++r) {
        int row = tm * 32 + i * 16 + r + (lhi << 3);
        float val = acc[i][j].f[r] + bn + ((float)row * (1.f / 2048.f)) * wz;
        if (!transposed)
          outRM[(size_t)b * FP * 512 + (size_t)row * 512 + gcol] = (h16)val;
        else
          outTR[(size_t)b * 512 * FP + (size_t)gcol * FP + row] = (h16)val;
      }
    }
  }
}

// ---------------------------------------------------------------------------
// scores = q . k^T per head, 2x2 tiles, K=64.
// ---------------------------------------------------------------------------
__global__ void scores_kernel(const h16* __restrict__ qh, const h16* __restrict__ kh,
                              float* __restrict__ attn) {
  int lane = threadIdx.x & 31, wave = threadIdx.x >> 5;
  long tile = (long)blockIdx.x * 8 + wave;
  int bh  = (int)(tile / MT_SCORE);
  int rem = (int)(tile % MT_SCORE);
  int b = bh >> 3, h = bh & 7;
  int tl = rem / M2T, ts = rem % M2T;

  const h16* ar0 = qh + ((size_t)b * FP + (size_t)(tl * 32 + (lane & 15))) * 512 + h * 64;
  const h16* ar1 = ar0 + (size_t)16 * 512;
  const h16* br0 = kh + ((size_t)b * FP + (size_t)(ts * 32 + (lane & 15))) * 512 + h * 64;
  const h16* br1 = br0 + (size_t)16 * 512;

  FragF acc[2][2];
#pragma unroll
  for (int i = 0; i < 2; ++i)
#pragma unroll
    for (int j = 0; j < 2; ++j)
#pragma unroll
      for (int r = 0; r < 8; ++r) acc[i][j].f[r] = 0.f;

#pragma unroll
  for (int k0 = 0; k0 < 64; k0 += 32) {
    v16h a0 = frag_ld_a(ar0, k0, lane);
    v16h a1 = frag_ld_a(ar1, k0, lane);
    v16h b0 = frag_ld_b(br0, k0, lane);
    v16h b1 = frag_ld_b(br1, k0, lane);
    acc[0][0].v = wmma_f16(a0, b0, acc[0][0].v);
    acc[0][1].v = wmma_f16(a0, b1, acc[0][1].v);
    acc[1][0].v = wmma_f16(a1, b0, acc[1][0].v);
    acc[1][1].v = wmma_f16(a1, b1, acc[1][1].v);
  }

  int lhi = lane >> 4, lcol = lane & 15;
#pragma unroll
  for (int j = 0; j < 2; ++j) {
    int s = ts * 32 + j * 16 + lcol;
    if (s < FREQ) {
#pragma unroll
      for (int i = 0; i < 2; ++i) {
#pragma unroll
        for (int r = 0; r < 8; ++r) {
          int l = tl * 32 + i * 16 + r + (lhi << 3);
          if (l < FREQ)
            attn[(((size_t)b * H_ + h) * FREQ + l) * FREQ + s] = acc[i][j].f[r];
        }
      }
    }
  }
}

// ---------------------------------------------------------------------------
__global__ void softmax_kernel(float* __restrict__ attn) {
  size_t row = blockIdx.x;
  float* p = attn + row * FREQ;
  const float scale = 0.125f;
  __shared__ float red[256];
  int t = threadIdx.x;

  float x[5], e[5];
  float m = -3.4e38f;
#pragma unroll
  for (int j = 0; j < 5; ++j) {
    int idx = t + j * 256;
    x[j] = (idx < FREQ) ? p[idx] * scale : -3.4e38f;
    m = fmaxf(m, x[j]);
    e[j] = 0.f;
  }
  red[t] = m;
  __syncthreads();
  for (int o = 128; o > 0; o >>= 1) {
    if (t < o) red[t] = fmaxf(red[t], red[t + o]);
    __syncthreads();
  }
  m = red[0];
  __syncthreads();

  float s = 0.f;
#pragma unroll
  for (int j = 0; j < 5; ++j) {
    int idx = t + j * 256;
    if (idx < FREQ) { e[j] = __expf(x[j] - m); s += e[j]; }
  }
  red[t] = s;
  __syncthreads();
  for (int o = 128; o > 0; o >>= 1) {
    if (t < o) red[t] += red[t + o];
    __syncthreads();
  }
  float inv = 1.f / red[0];
#pragma unroll
  for (int j = 0; j < 5; ++j) {
    int idx = t + j * 256;
    if (idx < FREQ) p[idx] = e[j] * inv;
  }
}

// ---------------------------------------------------------------------------
// ctx = P @ V, 2x2 tiles, K padded to 1056 (guarded at 1025).
// ---------------------------------------------------------------------------
__global__ void ctx_kernel(const float* __restrict__ attn, const h16* __restrict__ vhT,
                           h16* __restrict__ ctx) {
  int lane = threadIdx.x & 31, wave = threadIdx.x >> 5;
  long tile = (long)blockIdx.x * 8 + wave;
  int bh  = (int)(tile / MT_CTX);
  int rem = (int)(tile % MT_CTX);
  int b = bh >> 3, h = bh & 7;
  int tm = rem >> 1, tn = rem & 1;

  int l0 = tm * 32 + (lane & 15);
  int l1 = l0 + 16;
  bool mv0 = (l0 < FREQ), mv1 = (l1 < FREQ);
  const float* ar0 = attn + (((size_t)b * H_ + h) * FREQ + (mv0 ? l0 : 0)) * FREQ;
  const float* ar1 = attn + (((size_t)b * H_ + h) * FREQ + (mv1 ? l1 : 0)) * FREQ;
  const h16* br0 = vhT + ((size_t)b * 512 + (size_t)(h * 64 + tn * 32 + (lane & 15))) * FP;
  const h16* br1 = br0 + (size_t)16 * FP;

  FragF acc[2][2];
#pragma unroll
  for (int i = 0; i < 2; ++i)
#pragma unroll
    for (int j = 0; j < 2; ++j)
#pragma unroll
      for (int r = 0; r < 8; ++r) acc[i][j].f[r] = 0.f;

#pragma unroll 1
  for (int kt = 0; kt < 33; ++kt) {
    int k0 = kt * 32;
    v16h a0 = mv0 ? frag_ld_a_f32(ar0, k0, FREQ, lane) : frag_zero();
    v16h a1 = mv1 ? frag_ld_a_f32(ar1, k0, FREQ, lane) : frag_zero();
    v16h b0, b1;
    if (kt < 32) {
      b0 = frag_ld_b(br0, k0, lane);
      b1 = frag_ld_b(br1, k0, lane);
    } else {
      b0 = frag_ld_b_guard(br0, k0, FREQ, lane);
      b1 = frag_ld_b_guard(br1, k0, FREQ, lane);
    }
    acc[0][0].v = wmma_f16(a0, b0, acc[0][0].v);
    acc[0][1].v = wmma_f16(a0, b1, acc[0][1].v);
    acc[1][0].v = wmma_f16(a1, b0, acc[1][0].v);
    acc[1][1].v = wmma_f16(a1, b1, acc[1][1].v);
  }

  int lhi = lane >> 4, lcol = lane & 15;
#pragma unroll
  for (int i = 0; i < 2; ++i) {
#pragma unroll
    for (int j = 0; j < 2; ++j) {
#pragma unroll
      for (int r = 0; r < 8; ++r) {
        int row = tm * 32 + i * 16 + r + (lhi << 3);
        ctx[((size_t)b * FP + row) * 512 + h * 64 + tn * 32 + j * 16 + lcol] =
            (h16)acc[i][j].f[r];
      }
    }
  }
}

// ---------------------------------------------------------------------------
// Output projection with Tensor Data Mover: one block = 8 waves sharing a
// 32x512 f16 strip of Wo staged into LDS by TDM (tensor_load_to_lds + wait),
// then B fragments come from LDS (ds_load_b128) while A streams from global.
// ---------------------------------------------------------------------------
__global__ void outproj_kernel(const h16* __restrict__ ctx, const h16* __restrict__ Wo16,
                               const float* __restrict__ bo, float* __restrict__ y) {
  __shared__ __align__(16) h16 Bsh[32 * 512];   // 32 KB

  int lane = threadIdx.x & 31, wave = threadIdx.x >> 5;
  // grid: b (16) x tn (32) x mchunk (5)
  int blk = blockIdx.x;
  int b  = blk / (32 * OUT_MCHUNK);
  int r2 = blk % (32 * OUT_MCHUNK);
  int tn = r2 / OUT_MCHUNK;
  int mc = r2 % OUT_MCHUNK;
  int tm = mc * 8 + wave;

  const h16* wsrc = Wo16 + (size_t)tn * 32 * 512;   // contiguous 16384 f16

#if __has_builtin(__builtin_amdgcn_tensor_load_to_lds)
  if (threadIdx.x < 32) {
    unsigned long long ga = (unsigned long long)(uintptr_t)wsrc;
    unsigned ldsa = (unsigned)(uintptr_t)(&Bsh[0]);
    u32x4 g0;
    g0[0] = 1u;                                        // count=1 valid D#
    g0[1] = ldsa;                                      // lds_addr
    g0[2] = (unsigned)(ga & 0xffffffffu);              // global_addr[31:0]
    g0[3] = (unsigned)((ga >> 32) & 0x1ffffffu) | (2u << 30);  // [56:32], type=2
    i32x8 g1 = (i32x8)0;
    g1[0] = 1 << 16;                 // data_size = 1 (2-byte elements)
    g1[1] = (int)(0x4000u << 16);    // tensor_dim0 = 16384 (lo16 -> [31:16])
    g1[2] = 0;                       // tensor_dim0 hi / tensor_dim1 lo
    g1[3] = (int)(0x4000u << 16);    // tile_dim0 = 16384  ([127:112])
    g1[4] = 0;                       // tile_dim1 = 0 (1-D), tile_dim2 = 0
    g1[5] = 16384;                   // tensor_dim0_stride lo
    g1[6] = 0;
    g1[7] = 0;
    i32x4 gz = (i32x4)0;
#if defined(__clang_major__) && (__clang_major__ >= 23)
    i32x8 gz8 = (i32x8)0;
    __builtin_amdgcn_tensor_load_to_lds(g0, g1, gz, gz, gz8, 0);
#else
    __builtin_amdgcn_tensor_load_to_lds(g0, g1, gz, gz, 0);
#endif
    __builtin_amdgcn_s_wait_tensorcnt(0);
  }
#else
  for (int i = threadIdx.x; i < 32 * 512 / 8; i += 256)
    reinterpret_cast<float4*>(Bsh)[i] = reinterpret_cast<const float4*>(wsrc)[i];
#endif
  __syncthreads();

  if (tm < M2T) {
    const h16* ar0 = ctx + ((size_t)b * FP + (size_t)(tm * 32 + (lane & 15))) * 512;
    const h16* ar1 = ar0 + (size_t)16 * 512;
    const h16* br0 = &Bsh[(size_t)(lane & 15) * 512];        // LDS
    const h16* br1 = &Bsh[(size_t)(16 + (lane & 15)) * 512]; // LDS

    FragF acc[2][2];
#pragma unroll
    for (int i = 0; i < 2; ++i)
#pragma unroll
      for (int j = 0; j < 2; ++j)
#pragma unroll
        for (int r = 0; r < 8; ++r) acc[i][j].f[r] = 0.f;

#pragma unroll 4
    for (int k0 = 0; k0 < 512; k0 += 32) {
      if (k0 + 32 < 512) __builtin_prefetch(ar0 + k0 + 32, 0, 1);
      v16h a0 = frag_ld_a(ar0, k0, lane);
      v16h a1 = frag_ld_a(ar1, k0, lane);
      v16h b0 = frag_ld_b(br0, k0, lane);
      v16h b1 = frag_ld_b(br1, k0, lane);
      acc[0][0].v = wmma_f16(a0, b0, acc[0][0].v);
      acc[0][1].v = wmma_f16(a0, b1, acc[0][1].v);
      acc[1][0].v = wmma_f16(a1, b0, acc[1][0].v);
      acc[1][1].v = wmma_f16(a1, b1, acc[1][1].v);
    }

    int lhi = lane >> 4, lcol = lane & 15;
#pragma unroll
    for (int j = 0; j < 2; ++j) {
      int gcol = tn * 32 + j * 16 + lcol;
      float bn = bo[gcol];
#pragma unroll
      for (int i = 0; i < 2; ++i) {
#pragma unroll
        for (int r = 0; r < 8; ++r) {
          int row = tm * 32 + i * 16 + r + (lhi << 3);
          y[((size_t)b * FP + row) * 1024 + gcol] = acc[i][j].f[r] + bn;
        }
      }
    }
  }
}

// ---------------------------------------------------------------------------
__global__ void irfft_kernel(const float* __restrict__ y, float* __restrict__ out) {
  __shared__ float xr[1025], xi[1025];
  __shared__ float re[1024], im[1024];
  int bc = blockIdx.x;
  int b = bc >> 9, c = bc & 511;
  const float* yb = y + (size_t)b * FP * 1024 + 2 * c;
  for (int k = threadIdx.x; k <= 1024; k += 256) {
    xr[k] = yb[(size_t)k * 1024];
    xi[k] = yb[(size_t)k * 1024 + 1];
  }
  __syncthreads();
  for (int k = threadIdx.x; k < 1024; k += 256) {
    float Sr = xr[k] + xr[1024 - k], Si = xi[k] - xi[1024 - k];
    float Cr = xr[1024 - k] - xr[k], Ci = -xi[1024 - k] - xi[k];
    float th = 3.14159265359f * (float)k / 1024.f;
    float ct, st;
    __sincosf(th, &st, &ct);
    float tr = ct * Cr - st * Ci, ti = ct * Ci + st * Cr;
    float Dr = ti, Di = -tr;                 // D = -i * e^{+i th} * C
    int r = (int)(__brev((unsigned)k) >> 22);
    re[r] = 0.5f * (Sr + Dr);
    im[r] = 0.5f * (Si + Di);
  }
  __syncthreads();
  fft1024(re, im, 1.f);

  float* ob = out + (size_t)b * L_ * C_ + c;
  const float s = 1.f / 1024.f;
  for (int j = threadIdx.x; j < 1024; j += 256) {
    ob[(size_t)(2 * j) * C_]     = re[j] * s;
    ob[(size_t)(2 * j + 1) * C_] = im[j] * s;
  }
}

// ---------------------------------------------------------------------------
extern "C" void kernel_launch(void* const* d_in, const int* in_sizes, int n_in,
                              void* d_out, int out_size, void* d_ws, size_t ws_size,
                              hipStream_t stream) {
  (void)in_sizes; (void)n_in; (void)out_size; (void)ws_size;
  const float* q  = (const float*)d_in[0];
  const float* k  = (const float*)d_in[1];
  const float* v  = (const float*)d_in[2];
  const float* Wq = (const float*)d_in[3];
  const float* bq = (const float*)d_in[4];
  const float* Wk = (const float*)d_in[5];
  const float* bk = (const float*)d_in[6];
  const float* Wv = (const float*)d_in[7];
  const float* bv = (const float*)d_in[8];
  const float* Wo = (const float*)d_in[9];
  const float* bo = (const float*)d_in[10];

  float* out  = (float*)d_out;                    // (B, L, C)
  float* attn = out + (size_t)B_ * L_ * C_;       // (B, H, 1025, 1025)

  char* ws = (char*)d_ws;
  size_t off = 0;
  auto take = [&](size_t bytes) -> char* {
    char* p = ws + off;
    off += (bytes + 255) & ~(size_t)255;
    return p;
  };
  h16* Aq   = (h16*)take((size_t)B_ * FP * 1024 * 2);
  h16* Ak   = (h16*)take((size_t)B_ * FP * 1024 * 2);
  h16* Av   = (h16*)take((size_t)B_ * FP * 1024 * 2);
  h16* qh   = (h16*)take((size_t)B_ * FP * 512 * 2);
  h16* kh   = (h16*)take((size_t)B_ * FP * 512 * 2);
  h16* vhT  = (h16*)take((size_t)B_ * 512 * FP * 2);
  h16* ctx  = (h16*)take((size_t)B_ * FP * 512 * 2);
  h16* Wqc  = (h16*)take((size_t)512 * 1024 * 2);
  h16* Wkc  = (h16*)take((size_t)512 * 1024 * 2);
  h16* Wvc  = (h16*)take((size_t)512 * 1024 * 2);
  h16* Wo16 = (h16*)take((size_t)1024 * 512 * 2);
  float* wsq = (float*)take(512 * 4);
  float* wsk = (float*)take(512 * 4);
  float* wsv = (float*)take(512 * 4);
  float* ybuf = (float*)take((size_t)B_ * FP * 1024 * 4);

  prep_weights_kernel<<<512, 256, 0, stream>>>(Wq, Wk, Wv, Wo, Wqc, Wkc, Wvc,
                                               Wo16, wsq, wsk, wsv);
  rfft_feats_kernel<<<B_ * C_, 256, 0, stream>>>(q, Aq);
  rfft_feats_kernel<<<B_ * C_, 256, 0, stream>>>(k, Ak);
  rfft_feats_kernel<<<B_ * C_, 256, 0, stream>>>(v, Av);

  proj_gemm_kernel<<<(B_ * MT_PROJ) / 8, 256, 0, stream>>>(Aq, Wqc, bq, wsq, qh, vhT, 0);
  proj_gemm_kernel<<<(B_ * MT_PROJ) / 8, 256, 0, stream>>>(Ak, Wkc, bk, wsk, kh, vhT, 0);
  proj_gemm_kernel<<<(B_ * MT_PROJ) / 8, 256, 0, stream>>>(Av, Wvc, bv, wsv, qh, vhT, 1);

  scores_kernel<<<(B_ * H_ * MT_SCORE) / 8, 256, 0, stream>>>(qh, kh, attn);
  softmax_kernel<<<B_ * H_ * FREQ, 256, 0, stream>>>(attn);
  ctx_kernel<<<(B_ * H_ * MT_CTX) / 8, 256, 0, stream>>>(attn, vhT, ctx);
  outproj_kernel<<<B_ * 32 * OUT_MCHUNK, 256, 0, stream>>>(ctx, Wo16, bo, ybuf);
  irfft_kernel<<<B_ * C_, 256, 0, stream>>>(ybuf, out);
}
